// TRNCell_90640989815618
// MI455X (gfx1250) — matrix operand order repaired and checked by reading, not compiled, not run on hardware
//
#include <hip/hip_runtime.h>
#include <math.h>

// ---------------------------------------------------------------------------
// CDNA5 (gfx1250) WMMA GEMM-based TRN cell.
//   - bf16 A/B operands, f32 accumulation via v_wmma_f32_16x16x32_bf16
//   - weights + activations kept in bf16 -> BOTH tiles staged with
//     global_load_async_to_lds_b128 (ASYNCcnt), zero loadcnt stalls in loop
//   - double-buffered LDS, one barrier per K-step
// ---------------------------------------------------------------------------

typedef __attribute__((ext_vector_type(16))) __bf16 v16bf;
typedef __attribute__((ext_vector_type(8)))  __bf16 v8bf;
typedef __attribute__((ext_vector_type(8)))  float  v8f;

// exact parameter types of __builtin_amdgcn_global_load_async_to_lds_b128
typedef int async_i4 __attribute__((vector_size(4 * sizeof(int))));
typedef __attribute__((address_space(1))) async_i4* gas_i4p;
typedef __attribute__((address_space(3))) async_i4* las_i4p;

#define LDA_S 40   // 32 + 8 pad (bf16 elems) to dodge LDS bank conflicts
#define LDW_S 40

#if __has_builtin(__builtin_amdgcn_global_load_async_to_lds_b128)
#define ASYNC_LDS 1
#else
#define ASYNC_LDS 0
#endif

__device__ __forceinline__ void async_copy16(const __bf16* gsrc, __bf16* ldst) {
#if ASYNC_LDS
  __builtin_amdgcn_global_load_async_to_lds_b128(
      (gas_i4p)(void*)gsrc, (las_i4p)(void*)ldst, 0, 0);
#else
  *(v8bf*)ldst = *(const v8bf*)gsrc;
#endif
}

__device__ __forceinline__ void wait_async_zero() {
#if ASYNC_LDS
#if __has_builtin(__builtin_amdgcn_s_wait_asynccnt)
  __builtin_amdgcn_s_wait_asynccnt(0);
#else
  asm volatile("s_wait_asynccnt 0x0" ::: "memory");
#endif
#endif
}

// ---------------------------- small utility kernels ------------------------

__global__ void cvt_f32_bf16(const float* __restrict__ src,
                             __bf16* __restrict__ dst, int n) {
  int i = blockIdx.x * blockDim.x + threadIdx.x;
  if (i < n) dst[i] = (__bf16)src[i];
}

__global__ void cvt_scale_f32_bf16(const float* __restrict__ src,
                                   __bf16* __restrict__ dst, float scale,
                                   int n) {
  int i = blockIdx.x * blockDim.x + threadIdx.x;
  if (i < n) dst[i] = (__bf16)(src[i] * scale);
}

__global__ void fill_zero(float* __restrict__ dst, int n) {
  int i = blockIdx.x * blockDim.x + threadIdx.x;
  if (i < n) dst[i] = 0.0f;
}

// LSTM pointwise: gates [B,4H] (i,f,g,o), c_in [B,H] -> c_out (+h variants).
// h emitted as f32 (optional), bf16 (optional, feeds next GEMMs), and into a
// running-sum accumulator (optional, for the mean over the 32 steps).
__global__ void lstm_pointwise(const float* __restrict__ gates,
                               const float* __restrict__ c_in,
                               float* __restrict__ h_out,
                               __bf16* __restrict__ h_bf,
                               float* __restrict__ c_out,
                               float* __restrict__ h_accum,
                               int firstFlag, int H) {
  int idx = blockIdx.x * blockDim.x + threadIdx.x;   // over B*H (exact)
  int b = idx / H;
  int h = idx - b * H;
  const float* g = gates + (size_t)b * 4 * H;
  float ig = g[h];
  float fg = g[H + h];
  float gg = g[2 * H + h];
  float og = g[3 * H + h];
  float si = 1.0f / (1.0f + __expf(-ig));
  float sf = 1.0f / (1.0f + __expf(-fg));
  float so = 1.0f / (1.0f + __expf(-og));
  float tg = tanhf(gg);
  float c2 = sf * c_in[idx] + si * tg;
  float h2 = so * tanhf(c2);
  c_out[idx] = c2;
  if (h_out) h_out[idx] = h2;
  if (h_bf)  h_bf[idx]  = (__bf16)h2;
  if (h_accum) {
    if (firstFlag) h_accum[idx] = h2;
    else           h_accum[idx] += h2;
  }
}

// ---------------------------- WMMA GEMM kernel ------------------------------
// D[M,N] = act( A[M,K] x W[N,K]^T + bias0 + bias1 (+ C if accFlag) )
// A: bf16 row-major (lda). W: bf16 row-major (ldw).
// Outputs: C (f32, nullable) and/or Cbf (bf16, nullable), both ldc.
// Block tile 128x64, 8 wave32 (4x2), each wave 32x32 = 2x2 WMMA 16x16 tiles.
// K pipelined with double-buffered LDS; BOTH tiles staged via async LDS DMA.
__global__ __launch_bounds__(256) void gemm_wmma_bf16(
    const __bf16* __restrict__ A, int lda,
    const __bf16* __restrict__ W, int ldw,
    float* __restrict__ C, __bf16* __restrict__ Cbf, int ldc,
    int K,
    const float* __restrict__ bias0, const float* __restrict__ bias1,
    int accFlag, int reluFlag) {
  __shared__ __align__(16) __bf16 As[2][128 * LDA_S];
  __shared__ __align__(16) __bf16 Ws[2][64 * LDW_S];

  const int tid   = threadIdx.x;
  const int lane  = tid & 31;
  const int wave  = tid >> 5;
  const int waveM = wave >> 1;   // 0..3
  const int waveN = wave & 1;    // 0..1
  const int mBase = blockIdx.y * 128;
  const int nBase = blockIdx.x * 64;
  const int m16   = lane & 15;
  const int hi    = lane >> 4;   // 0 or 1

  // ---- tile staging: 3 async b128 per thread (2 for A, 1 for W) ----
  auto stage = [&](int buf, int kBase) {
    {  // W tile: 64 rows x 32 k  (256 thr x 16B)
      int row = tid >> 2;              // 0..63
      int c8  = tid & 3;
      async_copy16(W + (size_t)(nBase + row) * ldw + kBase + c8 * 8,
                   &Ws[buf][row * LDW_S + c8 * 8]);
    }
#pragma unroll
    for (int i = 0; i < 2; ++i) {  // A tile: 128 rows x 32 k (256 thr x 32B)
      int linear = tid + i * 256;      // 0..511 16B slots
      int row = linear >> 2;           // 4 slots per 32-wide row
      int c8  = linear & 3;
      async_copy16(A + (size_t)(mBase + row) * lda + kBase + c8 * 8,
                   &As[buf][row * LDA_S + c8 * 8]);
    }
  };

  v8f acc00 = {}, acc01 = {}, acc10 = {}, acc11 = {};

  const int nk = K >> 5;
  stage(0, 0);  // prologue: fill buffer 0
  for (int kb = 0; kb < nk; ++kb) {
    wait_async_zero();     // our async tile DMAs into LDS are done
    __syncthreads();       // everyone's tile visible

    if (kb + 1 < nk) {     // issue next tile into the other buffer;
      stage((kb + 1) & 1, (kb + 1) << 5);  // overlaps with WMMAs below
      __builtin_prefetch(A + (size_t)(mBase + (tid >> 1)) * lda + ((kb + 2) << 5), 0, 1);
    }

    const __bf16* Ab = &As[kb & 1][0];
    const __bf16* Wb = &Ws[kb & 1][0];

    // ---- gather fragments per ISA VGPR layouts ----
    // A 16x32: lane m=lane%16; lanes<16 hold K 0..7 & 16..23, lanes>=16 K+8
    v16bf a0, a1, b0, b1;
    {
      const __bf16* p0 = &Ab[(waveM * 32 +      m16) * LDA_S + hi * 8];
      const __bf16* p1 = &Ab[(waveM * 32 + 16 + m16) * LDA_S + hi * 8];
      v8bf l0 = *(const v8bf*)p0, h0 = *(const v8bf*)(p0 + 16);
      v8bf l1 = *(const v8bf*)p1, h1 = *(const v8bf*)(p1 + 16);
#pragma unroll
      for (int i = 0; i < 8; ++i) {
        a0[i] = l0[i]; a0[i + 8] = h0[i];
        a1[i] = l1[i]; a1[i + 8] = h1[i];
      }
    }
    // B 32x16: lane n=lane%16; lanes<16 hold K 0..7 & 16..23, lanes>=16 K+8
    {
      const __bf16* q0 = &Wb[(waveN * 32 +      m16) * LDW_S + hi * 8];
      const __bf16* q1 = &Wb[(waveN * 32 + 16 + m16) * LDW_S + hi * 8];
      v8bf l0 = *(const v8bf*)q0, h0 = *(const v8bf*)(q0 + 16);
      v8bf l1 = *(const v8bf*)q1, h1 = *(const v8bf*)(q1 + 16);
#pragma unroll
      for (int i = 0; i < 8; ++i) {
        b0[i] = l0[i]; b0[i + 8] = h0[i];
        b1[i] = l1[i]; b1[i + 8] = h1[i];
      }
    }
    acc00 = __builtin_amdgcn_wmma_f32_16x16x32_bf16(false, a0, false, b0,
                                                    (short)0, acc00, false, false);
    acc01 = __builtin_amdgcn_wmma_f32_16x16x32_bf16(false, a0, false, b1,
                                                    (short)0, acc01, false, false);
    acc10 = __builtin_amdgcn_wmma_f32_16x16x32_bf16(false, a1, false, b0,
                                                    (short)0, acc10, false, false);
    acc11 = __builtin_amdgcn_wmma_f32_16x16x32_bf16(false, a1, false, b1,
                                                    (short)0, acc11, false, false);
  }

  // ---- epilogue: C/D layout lane=col%16, VGPR v -> row v + 8*(lane>=16) ----
  auto store_tile = [&](const v8f& accv, int tm, int tn) {
    int gcol = nBase + waveN * 32 + tn * 16 + m16;
    float bsum = 0.0f;
    if (bias0) bsum += bias0[gcol];
    if (bias1) bsum += bias1[gcol];
#pragma unroll
    for (int v = 0; v < 8; ++v) {
      int grow = mBase + waveM * 32 + tm * 16 + v + hi * 8;
      size_t idx = (size_t)grow * ldc + gcol;
      float val = accv[v] + bsum;
      if (accFlag)  val += C[idx];
      if (reluFlag) val = val > 0.0f ? val : 0.0f;
      if (C)   C[idx]   = val;
      if (Cbf) Cbf[idx] = (__bf16)val;
    }
  };
  store_tile(acc00, 0, 0);
  store_tile(acc01, 0, 1);
  store_tile(acc10, 1, 0);
  store_tile(acc11, 1, 1);
}

// ------------------------------- host driver --------------------------------

extern "C" void kernel_launch(void* const* d_in, const int* in_sizes, int n_in,
                              void* d_out, int out_size, void* d_ws,
                              size_t ws_size, hipStream_t stream) {
  const int Bn = 1024, Din = 1024, H = 1024, Dout = 512, T = 32;

  const float* x        = (const float*)d_in[0];
  const float* hx       = (const float*)d_in[1];
  const float* cx       = (const float*)d_in[2];
  const float* W_ih_dec = (const float*)d_in[3];
  const float* W_hh_dec = (const float*)d_in[4];
  const float* b_ih_dec = (const float*)d_in[5];
  const float* b_hh_dec = (const float*)d_in[6];
  const float* W_r0     = (const float*)d_in[7];
  const float* b_r0     = (const float*)d_in[8];
  const float* W_rp     = (const float*)d_in[9];
  const float* b_rp     = (const float*)d_in[10];
  const float* W_rr     = (const float*)d_in[11];
  const float* b_rr     = (const float*)d_in[12];
  const float* W_s0     = (const float*)d_in[13];
  const float* b_s0     = (const float*)d_in[14];
  const float* W_ih_sta = (const float*)d_in[15];
  const float* W_hh_sta = (const float*)d_in[16];
  const float* b_ih_sta = (const float*)d_in[17];
  const float* b_hh_sta = (const float*)d_in[18];
  const float* W_sn     = (const float*)d_in[19];
  const float* b_sn     = (const float*)d_in[20];

  float* out    = (float*)d_out;
  float* Act    = out;                                 // [B, Dout]
  float* ActBar = out + (size_t)Bn * Dout;             // [T, B, Dout]
  float* hx2    = ActBar + (size_t)T * Bn * Dout;      // [B, H]
  float* cx2    = hx2 + (size_t)Bn * H;                // [B, H]

  // -------- workspace carve-up (256B aligned) --------
  char* ws = (char*)d_ws;
  size_t off = 0;
  auto carve = [&](size_t bytes) -> char* {
    char* p = ws + off;
    off = (off + bytes + 255) & ~(size_t)255;
    return p;
  };
  // bf16 weights
  __bf16* wIhDec = (__bf16*)carve((size_t)4 * H * Dout * 2);     // [4096,512]
  __bf16* wHhDec = (__bf16*)carve((size_t)4 * H * H * 2);        // [4096,1024]
  __bf16* wR0    = (__bf16*)carve((size_t)H * H * 2);
  __bf16* wRp    = (__bf16*)carve((size_t)Dout * H * 2);
  __bf16* wRr    = (__bf16*)carve((size_t)Dout * Dout * 2);
  __bf16* wS0    = (__bf16*)carve((size_t)Din * H * 2);
  __bf16* wIhSta = (__bf16*)carve((size_t)4 * H * 2 * Din * 2);  // [4096,2048]
  __bf16* wHhSta = (__bf16*)carve((size_t)4 * H * H * 2);
  __bf16* wSn    = (__bf16*)carve((size_t)Dout * H * 2);
  // f32 state
  float* gates = (float*)carve((size_t)Bn * 4 * H * 4);          // [B,4096]
  float* cd    = (float*)carve((size_t)Bn * H * 4);
  float* hbar  = (float*)carve((size_t)Bn * H * 4);
  // bf16 activations (GEMM A operands)
  __bf16* xBf    = (__bf16*)carve((size_t)Bn * Din * 2);
  __bf16* hxBf   = (__bf16*)carve((size_t)Bn * H * 2);
  __bf16* hdBf   = (__bf16*)carve((size_t)Bn * H * 2);
  __bf16* rBf    = (__bf16*)carve((size_t)Bn * Dout * 2);
  __bf16* pBf    = (__bf16*)carve((size_t)Bn * Dout * 2);
  __bf16* hbarBf = (__bf16*)carve((size_t)Bn * H * 2);
  __bf16* xbarBf = (__bf16*)carve((size_t)Bn * Din * 2);
  __bf16* hx2Bf  = (__bf16*)carve((size_t)Bn * H * 2);
  (void)ws_size; (void)in_sizes; (void)n_in; (void)out_size;

  auto cvt = [&](const float* s, __bf16* d, int n) {
    cvt_f32_bf16<<<(n + 255) / 256, 256, 0, stream>>>(s, d, n);
  };
  auto gemm = [&](const __bf16* A, int lda, const __bf16* Wm, int ldw,
                  float* Cm, __bf16* Cb, int ldc, int M, int N, int K,
                  const float* bb0, const float* bb1, int accF, int reluF) {
    dim3 grid(N / 64, M / 128);
    gemm_wmma_bf16<<<grid, 256, 0, stream>>>(A, lda, Wm, ldw, Cm, Cb, ldc, K,
                                             bb0, bb1, accF, reluF);
  };

  // 1) weights + input activations -> bf16 (resident in L2 thereafter)
  cvt(W_ih_dec, wIhDec, 4 * H * Dout);
  cvt(W_hh_dec, wHhDec, 4 * H * H);
  cvt(W_r0,     wR0,    H * H);
  cvt(W_rp,     wRp,    Dout * H);
  cvt(W_rr,     wRr,    Dout * Dout);
  cvt(W_s0,     wS0,    Din * H);
  cvt(W_ih_sta, wIhSta, 4 * H * 2 * Din);
  cvt(W_hh_sta, wHhSta, 4 * H * H);
  cvt(W_sn,     wSn,    Dout * H);
  cvt(x,  xBf,  Bn * Din);
  cvt(hx, hxBf, Bn * H);

  // 2) init: r0 = 0 (bf16 zeros), hd0 = hx @ W_r0^T + b_r0 (bf16), cd = cx
  fill_zero<<<(Bn * Dout / 2 + 255) / 256, 256, 0, stream>>>((float*)rBf,
                                                             Bn * Dout / 2);
  gemm(hxBf, H, wR0, H, nullptr, hdBf, H, Bn, H, H, b_r0, nullptr, 0, 0);
  (void)hipMemcpyAsync(cd, cx, (size_t)Bn * H * 4, hipMemcpyDeviceToDevice,
                       stream);

  // 3) 32-step decoder loop (sequential dependency chain)
  for (int t = 0; t < T; ++t) {
    // gates = r @ W_ih_dec^T + (b_ih + b_hh)
    gemm(rBf, Dout, wIhDec, Dout, gates, nullptr, 4 * H, Bn, 4 * H, Dout,
         b_ih_dec, b_hh_dec, 0, 0);
    // gates += hd @ W_hh_dec^T
    gemm(hdBf, H, wHhDec, H, gates, nullptr, 4 * H, Bn, 4 * H, H, nullptr,
         nullptr, 1, 0);
    // (hd, cd) = LSTM(gates, cd); hbar (+)= hd; hd kept as bf16 for GEMMs
    lstm_pointwise<<<(Bn * H) / 256, 256, 0, stream>>>(
        gates, cd, nullptr, hdBf, cd, hbar, (t == 0) ? 1 : 0, H);
    // p = hd @ W_rp^T + b_rp  -> Act_bar[t] (f32 output) + bf16 copy
    float* pOut = ActBar + (size_t)t * Bn * Dout;
    gemm(hdBf, H, wRp, H, pOut, pBf, Dout, Bn, Dout, H, b_rp, nullptr, 0, 0);
    // r = p @ W_rr^T + b_rr   (bf16 only; feeds next step's GEMM)
    gemm(pBf, Dout, wRr, Dout, nullptr, rBf, Dout, Bn, Dout, Dout, b_rr,
         nullptr, 0, 0);
  }

  // 4) xbar = relu( (hbar/32) @ W_s0^T + b_s0 )   (mean folded into convert)
  cvt_scale_f32_bf16<<<(Bn * H + 255) / 256, 256, 0, stream>>>(
      hbar, hbarBf, 1.0f / 32.0f, Bn * H);
  gemm(hbarBf, H, wS0, H, nullptr, xbarBf, Din, Bn, Din, H, b_s0, nullptr, 0,
       1);

  // 5) sta LSTM: gates = [x, xbar] @ W_ih_sta^T + b's + hx @ W_hh_sta^T
  gemm(xBf, Din, wIhSta, 2 * Din, gates, nullptr, 4 * H, Bn, 4 * H, Din,
       b_ih_sta, b_hh_sta, 0, 0);                       // x part (cols 0..1023)
  gemm(xbarBf, Din, wIhSta + Din, 2 * Din, gates, nullptr, 4 * H, Bn, 4 * H,
       Din, nullptr, nullptr, 1, 0);                    // xbar part
  gemm(hxBf, H, wHhSta, H, gates, nullptr, 4 * H, Bn, 4 * H, H, nullptr,
       nullptr, 1, 0);
  lstm_pointwise<<<(Bn * H) / 256, 256, 0, stream>>>(
      gates, cx, hx2, hx2Bf, cx2, nullptr, 0, H);

  // 6) Act = hx2 @ W_sn^T + b_sn
  gemm(hx2Bf, H, wSn, H, Act, nullptr, Dout, Bn, Dout, H, b_sn, nullptr, 0, 0);
}
